// MultiheadAttention_29884382446396
// MI455X (gfx1250) — matrix operand order repaired and checked by reading, not compiled, not run on hardware
//
#include <hip/hip_runtime.h>
#include <cstdint>
#include <cstddef>

#define BATCH 2
#define SEQ   2048
#define EMB   1024
#define NH    16
#define DH    64

typedef unsigned short u16b;
typedef __attribute__((ext_vector_type(16))) __bf16 v16bf;
typedef __attribute__((ext_vector_type(8)))  float  v8f;

struct Frag16 { uint32_t u[8]; };   // 16 bf16 operand fragment (8 VGPRs)

__device__ __forceinline__ u16b f2bf(float f) {
  uint32_t u = __builtin_bit_cast(uint32_t, f);
  u += 0x7fffu + ((u >> 16) & 1u);          // round-to-nearest-even
  return (u16b)(u >> 16);
}

__device__ __forceinline__ v8f wmma_bf16(const Frag16 &a, const Frag16 &b, v8f c) {
  // D = A(16x32 bf16) * B(32x16 bf16) + C(f32)
  return __builtin_amdgcn_wmma_f32_16x16x32_bf16(
      false, __builtin_bit_cast(v16bf, a),
      false, __builtin_bit_cast(v16bf, b),
      (short)0, c, false, false);
}

// CDNA5 async global->LDS copy (16B per lane), tracked by ASYNCcnt.
__device__ __forceinline__ void async_ld_b128(uint32_t lds_byte_off, const void* gptr) {
  asm volatile("global_load_async_to_lds_b128 %0, %1, off"
               :: "v"(lds_byte_off), "v"(gptr) : "memory");
}
__device__ __forceinline__ void wait_async0() {
  asm volatile("s_wait_asynccnt 0x0" ::: "memory");
}
__device__ __forceinline__ uint32_t lds_off(const void* p) {
  return (uint32_t)(uintptr_t)p;   // LDS aperture: addr[31:0] = LDS byte offset
}

// Row reductions across the 16 lanes of a half-wave (C-matrix rows live there).
__device__ __forceinline__ float rmax16(float v) {
#pragma unroll
  for (int m = 1; m < 16; m <<= 1) v = fmaxf(v, __shfl_xor(v, m, 32));
  return v;
}
__device__ __forceinline__ float rsum16(float v) {
#pragma unroll
  for (int m = 1; m < 16; m <<= 1) v += __shfl_xor(v, m, 32);
  return v;
}

// ---------------------------------------------------------------- fp32 -> bf16
__global__ __launch_bounds__(256) void k_f32_to_bf16(const float* __restrict__ s,
                                                     u16b* __restrict__ d, int n) {
  for (int i = blockIdx.x * 256 + threadIdx.x; i < n; i += gridDim.x * 256)
    d[i] = f2bf(s[i]);
}

// ---------------------------------------------------------------- LayerNorm
__global__ __launch_bounds__(256) void k_layernorm(const float* __restrict__ x,
                                                   const float* __restrict__ g,
                                                   const float* __restrict__ b,
                                                   u16b* __restrict__ xn) {
  const int row = blockIdx.x;
  const float* xr = x + (size_t)row * EMB;
  const int tid = threadIdx.x, lane = tid & 31, wid = tid >> 5;
  float v[4], s = 0.f, s2 = 0.f;
#pragma unroll
  for (int i = 0; i < 4; ++i) { v[i] = xr[tid + i * 256]; s += v[i]; s2 += v[i] * v[i]; }
#pragma unroll
  for (int m = 1; m < 32; m <<= 1) { s += __shfl_xor(s, m, 32); s2 += __shfl_xor(s2, m, 32); }
  __shared__ float ps[8], ps2[8];
  if (lane == 0) { ps[wid] = s; ps2[wid] = s2; }
  __syncthreads();
  float S = 0.f, S2 = 0.f;
#pragma unroll
  for (int w = 0; w < 8; ++w) { S += ps[w]; S2 += ps2[w]; }
  const float mu   = S * (1.f / EMB);
  const float var  = S2 * (1.f / EMB) - mu * mu;
  const float rstd = rsqrtf(var + 1e-5f);
#pragma unroll
  for (int i = 0; i < 4; ++i) {
    const int j = tid + i * 256;
    xn[(size_t)row * EMB + j] = f2bf((v[i] - mu) * rstd * g[j] + b[j]);
  }
}

// ------------------------------------------------- NT GEMM: C = A * W^T + bias
// A[M,K] bf16 (K contiguous), W[N,K] bf16 (K contiguous), C[M,N] f32.
// Block tile 64x128, BK=32; 8 waves in 2x4; wave tile 32x32 = 2x2 WMMA frags.
// Double-buffered LDS filled by async global->LDS copies (ASYNCcnt).
__global__ __launch_bounds__(256) void k_gemm_bf16_nt(const u16b* __restrict__ A,
                                                      const u16b* __restrict__ W,
                                                      const float* __restrict__ bias,
                                                      float* __restrict__ C,
                                                      int M, int N, int K) {
  __shared__ alignas(16) u16b Ash[2][64 * 32];
  __shared__ alignas(16) u16b Wsh[2][128 * 32];
  const int tid = threadIdx.x, lane = tid & 31, wid = tid >> 5;
  const int ln = lane & 15, hi = lane >> 4;
  const int m0 = blockIdx.y * 64, n0 = blockIdx.x * 128;
  const int wm = wid >> 2, wn = wid & 3;

  const int arow = tid >> 2, ac8 = (tid & 3) * 8;   // A: 64x32 = 256 x 16B chunks
  auto issue = [&](int k0, int buf) {
    async_ld_b128(lds_off(&Ash[buf][arow * 32 + ac8]),
                  A + (size_t)(m0 + arow) * K + k0 + ac8);
#pragma unroll
    for (int rep = 0; rep < 2; ++rep) {             // W: 128x32 = 512 x 16B chunks
      const int id = tid + rep * 256;
      const int wrow = id >> 2, wc8 = (id & 3) * 8;
      async_ld_b128(lds_off(&Wsh[buf][wrow * 32 + wc8]),
                    W + (size_t)(n0 + wrow) * K + k0 + wc8);
    }
  };

  v8f c[2][2];
#pragma unroll
  for (int mi = 0; mi < 2; ++mi)
#pragma unroll
    for (int ni = 0; ni < 2; ++ni)
#pragma unroll
      for (int r = 0; r < 8; ++r) c[mi][ni][r] = 0.f;

  issue(0, 0);
  const int KT = K / 32;
  for (int kt = 0; kt < KT; ++kt) {
    const int buf = kt & 1;
    wait_async0();        // my wave's copies into buf are done
    __syncthreads();      // everyone's copies done; prior reads of buf^1 done
    if (kt + 1 < KT) issue((kt + 1) * 32, buf ^ 1);

    Frag16 a[2], bb[2];
#pragma unroll
    for (int mi = 0; mi < 2; ++mi) {               // A frag: lane = row
      const int r0 = (wm * 32 + mi * 16 + ln) * 32;
#pragma unroll
      for (int p = 0; p < 8; ++p) {
        const int kd = hi * 16 + 2 * p;
        a[mi].u[p] = *(const uint32_t*)&Ash[buf][r0 + kd];
      }
    }
#pragma unroll
    for (int ni = 0; ni < 2; ++ni) {               // B frag: lane = column (= W row)
      const int r0 = (wn * 32 + ni * 16 + ln) * 32 + hi * 16;
#pragma unroll
      for (int p = 0; p < 8; ++p)
        bb[ni].u[p] = *(const uint32_t*)&Wsh[buf][r0 + 2 * p];
    }
#pragma unroll
    for (int mi = 0; mi < 2; ++mi)
#pragma unroll
      for (int ni = 0; ni < 2; ++ni)
        c[mi][ni] = wmma_bf16(a[mi], bb[ni], c[mi][ni]);
  }
#pragma unroll
  for (int mi = 0; mi < 2; ++mi)
#pragma unroll
    for (int ni = 0; ni < 2; ++ni) {
      const int n = n0 + wn * 32 + ni * 16 + ln;
      const float bv = bias[n];
#pragma unroll
      for (int r = 0; r < 8; ++r) {
        const int m = m0 + wm * 32 + mi * 16 + r + 8 * hi;
        C[(size_t)m * N + n] = c[mi][ni][r] + bv;
      }
    }
}

// --------------------------------- RoPE + head split + per-head L2 norm/scale
__global__ __launch_bounds__(256) void k_rope_norm_split(const float* __restrict__ qkv,
                                                         const float* __restrict__ inv_freq,
                                                         const float* __restrict__ qknorm_scale,
                                                         u16b* __restrict__ Qo,
                                                         u16b* __restrict__ Ko,
                                                         u16b* __restrict__ Vo) {
  const int row = blockIdx.x;                 // b*SEQ + s
  const int b = row >> 11, s = row & 2047;
  const float* base = qkv + (size_t)row * (3 * EMB);
  __shared__ float qsh[EMB], ksh[EMB], rq[EMB], rk[EMB];
  __shared__ float hq[NH], hk[NH];
  const int tid = threadIdx.x;
  if (tid < NH) { hq[tid] = 0.f; hk[tid] = 0.f; }
#pragma unroll
  for (int i = 0; i < 4; ++i) {
    const int j = tid + i * 256;
    qsh[j] = base[j];
    ksh[j] = base[EMB + j];
    const int h = j >> 6, d = j & 63;
    Vo[((size_t)(b * NH + h) * SEQ + s) * DH + d] = f2bf(base[2 * EMB + j]);
  }
  __syncthreads();
#pragma unroll
  for (int i = 0; i < 4; ++i) {
    const int j = tid + i * 256;
    const float ang = (float)SEQ * inv_freq[j];   // position-independent in ref
    const float sv = sinf(ang), cv = cosf(ang);
    const float rotq = (j < EMB / 2) ? -qsh[2 * j + 1] : qsh[2 * (j - EMB / 2)];
    const float rotk = (j < EMB / 2) ? -ksh[2 * j + 1] : ksh[2 * (j - EMB / 2)];
    const float vq = qsh[j] * cv + rotq * sv;
    const float vk = ksh[j] * cv + rotk * sv;
    rq[j] = vq; rk[j] = vk;
    atomicAdd(&hq[j >> 6], vq * vq);
    atomicAdd(&hk[j >> 6], vk * vk);
  }
  __syncthreads();
  const float sc = qknorm_scale[0];
#pragma unroll
  for (int i = 0; i < 4; ++i) {
    const int j = tid + i * 256;
    const int h = j >> 6, d = j & 63;
    const float qs = sc / fmaxf(sqrtf(hq[h]), 1e-12f);
    const float ks = sc / fmaxf(sqrtf(hk[h]), 1e-12f);
    const size_t o = ((size_t)(b * NH + h) * SEQ + s) * DH + d;
    Qo[o] = f2bf(rq[j] * qs);
    Ko[o] = f2bf(rk[j] * ks);
  }
}

// --------------------------------------------- causal flash attention (WMMA)
// grid = B*H*(S/128); block = 8 waves; wave owns 16 q-rows x D=64 output.
__global__ __launch_bounds__(256) void k_flash_attn(const u16b* __restrict__ Q,
                                                    const u16b* __restrict__ Km,
                                                    const u16b* __restrict__ Vm,
                                                    u16b* __restrict__ O) {
  const int blk = blockIdx.x;
  const int qb = blk & 15, h = (blk >> 4) & 15, b = blk >> 8;
  const int q0 = qb * 128;
  const size_t head = (size_t)(b * NH + h) * SEQ * DH;
  const u16b* qbase = Q + head;
  const u16b* kbase = Km + head;
  const u16b* vbase = Vm + head;

  __shared__ alignas(16) u16b Ksh[32 * DH];      // [key][d]
  __shared__ alignas(16) u16b Vsh[DH * 32];      // transposed: [d][key]
  __shared__ alignas(16) u16b Psh[8][16 * 32];   // per-wave P repack

  const int tid = threadIdx.x, lane = tid & 31, wid = tid >> 5;
  const int ln = lane & 15, hi = lane >> 4;

  Frag16 qf[2];                                  // Q as A-frags, K-dim 0..63
  {
    const u16b* qr = qbase + (size_t)(q0 + wid * 16 + ln) * DH;
#pragma unroll
    for (int kk = 0; kk < 2; ++kk)
#pragma unroll
      for (int p = 0; p < 8; ++p) {
        const int kd = kk * 32 + hi * 16 + 2 * p;
        qf[kk].u[p] = *(const uint32_t*)(qr + kd);
      }
  }

  float m_r[8], l_r[8];
  v8f o[4];
#pragma unroll
  for (int r = 0; r < 8; ++r) { m_r[r] = -1e30f; l_r[r] = 0.f; }
#pragma unroll
  for (int nt = 0; nt < 4; ++nt)
#pragma unroll
    for (int r = 0; r < 8; ++r) o[nt][r] = 0.f;

  const int nkt = (q0 + 128) / 32;               // uniform per block (causal)
  for (int kt = 0; kt < nkt; ++kt) {
    __syncthreads();
    {
      const int krow = tid >> 3, c8 = (tid & 7) * 8;
      // K tile: async global->LDS (256 x 16B chunks)
      async_ld_b128(lds_off(&Ksh[krow * DH + c8]),
                    kbase + (size_t)(kt * 32 + krow) * DH + c8);
      // V tile: manual transpose into [d][key]
      const u16b* vp = vbase + (size_t)(kt * 32 + krow) * DH + c8;
#pragma unroll
      for (int e = 0; e < 4; ++e) {
        const uint32_t w = *(const uint32_t*)(vp + 2 * e);
        Vsh[(c8 + 2 * e) * 32 + krow]     = (u16b)(w & 0xffffu);
        Vsh[(c8 + 2 * e + 1) * 32 + krow] = (u16b)(w >> 16);
      }
    }
    wait_async0();
    __syncthreads();

    v8f sl[2];                                   // logits 16 x 32
#pragma unroll
    for (int nt = 0; nt < 2; ++nt) {
#pragma unroll
      for (int r = 0; r < 8; ++r) sl[nt][r] = 0.f;
#pragma unroll
      for (int kk = 0; kk < 2; ++kk) {
        Frag16 kf;                               // B = K^T: lane = key column
        const int r0 = (nt * 16 + ln) * DH + kk * 32 + hi * 16;
#pragma unroll
        for (int p = 0; p < 8; ++p) kf.u[p] = *(const uint32_t*)&Ksh[r0 + 2 * p];
        sl[nt] = wmma_bf16(qf[kk], kf, sl[nt]);
      }
    }

#pragma unroll
    for (int r = 0; r < 8; ++r) {                // online softmax per q-row
      const int qg = q0 + wid * 16 + r + 8 * hi;
#pragma unroll
      for (int nt = 0; nt < 2; ++nt) {
        const int kg = kt * 32 + nt * 16 + ln;
        if (kg > qg) sl[nt][r] = -1e30f;         // causal mask
      }
      const float mx   = rmax16(fmaxf(sl[0][r], sl[1][r]));
      const float mnew = fmaxf(m_r[r], mx);
      const float alpha = __expf(m_r[r] - mnew);
      const float p0 = __expf(sl[0][r] - mnew);
      const float p1 = __expf(sl[1][r] - mnew);
      const float rs = rsum16(p0 + p1);
      l_r[r] = l_r[r] * alpha + rs;
      m_r[r] = mnew;
#pragma unroll
      for (int nt = 0; nt < 4; ++nt) o[nt][r] *= alpha;
      Psh[wid][(r + 8 * hi) * 32 + ln]      = f2bf(p0);
      Psh[wid][(r + 8 * hi) * 32 + 16 + ln] = f2bf(p1);
    }
    __syncthreads();

    Frag16 pf;                                   // P as A-frag (16x32)
#pragma unroll
    for (int p = 0; p < 8; ++p) {
      const int kd = hi * 16 + 2 * p;
      pf.u[p] = *(const uint32_t*)&Psh[wid][ln * 32 + kd];
    }
#pragma unroll
    for (int nt = 0; nt < 4; ++nt) {             // O += P * V
      Frag16 vf;                                 // B = V: lane = d column
      const int r0 = (nt * 16 + ln) * 32 + hi * 16;
#pragma unroll
      for (int p = 0; p < 8; ++p) vf.u[p] = *(const uint32_t*)&Vsh[r0 + 2 * p];
      o[nt] = wmma_bf16(pf, vf, o[nt]);
    }
  }

#pragma unroll
  for (int r = 0; r < 8; ++r) {
    const float inv = 1.f / l_r[r];
    const int s = q0 + wid * 16 + r + 8 * hi;
#pragma unroll
    for (int nt = 0; nt < 4; ++nt)
      O[((size_t)(b * SEQ + s)) * EMB + h * DH + nt * 16 + ln] = f2bf(o[nt][r] * inv);
  }
}

// ---------------------------------------------------------------- launcher
extern "C" void kernel_launch(void* const* d_in, const int* in_sizes, int n_in,
                              void* d_out, int out_size, void* d_ws, size_t ws_size,
                              hipStream_t stream) {
  (void)in_sizes; (void)n_in; (void)out_size; (void)ws_size;
  const float* x      = (const float*)d_in[0];
  const float* ln_w   = (const float*)d_in[1];
  const float* ln_b   = (const float*)d_in[2];
  const float* qkv_w  = (const float*)d_in[3];
  const float* qkv_b  = (const float*)d_in[4];
  const float* qknorm = (const float*)d_in[5];
  const float* out_w  = (const float*)d_in[6];
  const float* out_b  = (const float*)d_in[7];
  const float* inv_f  = (const float*)d_in[8];

  char* ws = (char*)d_ws;
  size_t off = 0;
  auto take = [&](size_t bytes) {
    char* p = ws + off;
    off = (off + bytes + 255) & ~(size_t)255;
    return p;
  };
  u16b*  wqkv_bf = (u16b*)take((size_t)3 * EMB * EMB * 2);
  u16b*  wout_bf = (u16b*)take((size_t)EMB * EMB * 2);
  u16b*  xn      = (u16b*)take((size_t)BATCH * SEQ * EMB * 2);
  float* qkv     = (float*)take((size_t)BATCH * SEQ * 3 * EMB * 4);
  u16b*  qh      = (u16b*)take((size_t)BATCH * NH * SEQ * DH * 2);
  u16b*  kh      = (u16b*)take((size_t)BATCH * NH * SEQ * DH * 2);
  u16b*  vh      = (u16b*)take((size_t)BATCH * NH * SEQ * DH * 2);
  u16b*  attn    = (u16b*)take((size_t)BATCH * SEQ * EMB * 2);

  k_f32_to_bf16<<<2048, 256, 0, stream>>>(qkv_w, wqkv_bf, 3 * EMB * EMB);
  k_f32_to_bf16<<<1024, 256, 0, stream>>>(out_w, wout_bf, EMB * EMB);
  k_layernorm<<<BATCH * SEQ, 256, 0, stream>>>(x, ln_w, ln_b, xn);
  k_gemm_bf16_nt<<<dim3(3 * EMB / 128, BATCH * SEQ / 64), 256, 0, stream>>>(
      xn, wqkv_bf, qkv_b, qkv, BATCH * SEQ, 3 * EMB, EMB);
  k_rope_norm_split<<<BATCH * SEQ, 256, 0, stream>>>(qkv, inv_f, qknorm, qh, kh, vh);
  k_flash_attn<<<BATCH * NH * (SEQ / 128), 256, 0, stream>>>(qh, kh, vh, attn);
  k_gemm_bf16_nt<<<dim3(EMB / 128, BATCH * SEQ / 64), 256, 0, stream>>>(
      attn, wout_bf, out_b, (float*)d_out, BATCH * SEQ, EMB, EMB);
}
